// encoder_layer_14946486190133
// MI455X (gfx1250) — matrix operand order repaired and checked by reading, not compiled, not run on hardware
//
#include <hip/hip_runtime.h>
#include <cstdint>
#include <cstddef>

typedef __attribute__((ext_vector_type(16))) _Float16 v16h;
typedef __attribute__((ext_vector_type(8)))  float    v8f;

#define B_DIM 128
#define I_DIM 1024
#define N_DIM 2048
#define T_DIM 128

#define DECAY_F            0.77880078307140487f   // exp(-1/4)
#define ONE_MINUS_DECAY_F  0.22119921692859513f

#define MT 64      // t-tile per block
#define NT 128     // n-tile per block
#define KT 32      // k step (WMMA K for f16)
#define APITCH 34  // halves per As row (even -> 4B aligned pairs)
#define BPITCH 34  // halves per Bs row
#define CPITCH 68  // floats per Ct row (stride%64==4 -> conflict-light)

// Kernel 1: drive[b][n][t] = (x[b,:,t] . W[n,:] + bias[n]) * gain[n] * (1-decay)
// WMMA f16 inputs, f32 accumulate. Each wave owns a 32t x 32n subtile
// (2 A-frags x 2 B-frags -> 4 v_wmma per K-step). Output in final [B,N,T] layout.
__global__ __launch_bounds__(256) void snn_gemm_drive(
    const float* __restrict__ x,    // [B, I, T]
    const float* __restrict__ W,    // [N, I]
    const float* __restrict__ bias, // [N]
    const float* __restrict__ gain, // [N]
    float* __restrict__ out)        // [B, N, T] (drive -> spikes region)
{
    __shared__ __align__(16) unsigned char smem[NT * CPITCH * 4]; // 34816 B
    _Float16* As = reinterpret_cast<_Float16*>(smem);                   // [MT][APITCH] 4352 B
    _Float16* Bs = reinterpret_cast<_Float16*>(smem + MT * APITCH * 2); // [NT][BPITCH] 8704 B

    const int n0   = blockIdx.x * NT;
    const int t0   = blockIdx.y * MT;
    const int b    = blockIdx.z;
    const int tid  = threadIdx.x;
    const int lane = tid & 31;
    const int wave = tid >> 5;
    const int wm   = wave & 1;   // t offset wm*32
    const int wn   = wave >> 1;  // n offset wn*32
    const int h    = lane >> 4;  // lane half
    const int l16  = lane & 15;

    const float* xb = x + (size_t)b * I_DIM * T_DIM + t0; // index: + i*T + t

    v8f acc00 = {0.f,0.f,0.f,0.f,0.f,0.f,0.f,0.f};
    v8f acc01 = {0.f,0.f,0.f,0.f,0.f,0.f,0.f,0.f};
    v8f acc10 = {0.f,0.f,0.f,0.f,0.f,0.f,0.f,0.f};
    v8f acc11 = {0.f,0.f,0.f,0.f,0.f,0.f,0.f,0.f};

    for (int k0 = 0; k0 < I_DIM; k0 += KT) {
        // Prefetch next K tile (emits global_prefetch_b8)
        if (k0 + KT < I_DIM) {
            __builtin_prefetch(xb + (size_t)(k0 + KT + (tid >> 4)) * T_DIM + (tid & 15) * 4, 0, 1);
            __builtin_prefetch(W + (size_t)(n0 + (tid >> 3)) * I_DIM + (k0 + KT) + (tid & 7) * 4, 0, 1);
        }

        // Stage A: x[b, k0..k0+31, t0..t0+63] -> As[t][k] (f16, transposed)
        {
            const int tg = (tid & 15) * 4; // 4 consecutive t per thread
            const int kr = tid >> 4;       // k row 0..15 (+16 on rep)
            #pragma unroll
            for (int rep = 0; rep < 2; ++rep) {
                const int k = kr + rep * 16;
                const float4 vx = *(const float4*)(xb + (size_t)(k0 + k) * T_DIM + tg);
                As[(tg + 0) * APITCH + k] = (_Float16)vx.x;
                As[(tg + 1) * APITCH + k] = (_Float16)vx.y;
                As[(tg + 2) * APITCH + k] = (_Float16)vx.z;
                As[(tg + 3) * APITCH + k] = (_Float16)vx.w;
            }
        }
        // Stage B: W[n0..n0+127, k0..k0+31] -> Bs[n][k] (f16)
        {
            const int kg = (tid & 7) * 4;
            const int nr = tid >> 3; // 0..31 (+32 per rep)
            #pragma unroll
            for (int rep = 0; rep < 4; ++rep) {
                const int n = nr + rep * 32;
                const float4 vw = *(const float4*)(W + (size_t)(n0 + n) * I_DIM + k0 + kg);
                Bs[n * BPITCH + kg + 0] = (_Float16)vw.x;
                Bs[n * BPITCH + kg + 1] = (_Float16)vw.y;
                Bs[n * BPITCH + kg + 2] = (_Float16)vw.z;
                Bs[n * BPITCH + kg + 3] = (_Float16)vw.w;
            }
        }
        __syncthreads();

        // Build fragments per CDNA5 16-bit WMMA VGPR layout
        const _Float16* Arow0 = As + (wm * 32 + l16) * APITCH;
        const _Float16* Arow1 = Arow0 + 16 * APITCH;
        const _Float16* Brow0 = Bs + (wn * 32 + l16) * BPITCH;
        const _Float16* Brow1 = Brow0 + 16 * BPITCH;
        v16h af0, af1, bf0, bf1;
        #pragma unroll
        for (int v = 0; v < 8; ++v) {
            // A 16x32: vgpr v<4 -> K = 8h + 2v..; v>=4 -> K = 16 + 8h + 2(v-4)..
            const int ka = ((v & 4) << 2) + (h << 3) + ((v & 3) << 1);
            af0[2 * v]     = Arow0[ka];
            af0[2 * v + 1] = Arow0[ka + 1];
            af1[2 * v]     = Arow1[ka];
            af1[2 * v + 1] = Arow1[ka + 1];
            // B 32x16: vgpr v<4 -> K = 8h + 2v..; v>=4 -> K = 16 + 8h + 2(v-4)..
            const int kb = ((v & 4) << 2) + (h << 3) + ((v & 3) << 1);
            bf0[2 * v]     = Brow0[kb];
            bf0[2 * v + 1] = Brow0[kb + 1];
            bf1[2 * v]     = Brow1[kb];
            bf1[2 * v + 1] = Brow1[kb + 1];
        }

        acc00 = __builtin_amdgcn_wmma_f32_16x16x32_f16(false, af0, false, bf0,
                                                       (short)0, acc00, false, false);
        acc01 = __builtin_amdgcn_wmma_f32_16x16x32_f16(false, af0, false, bf1,
                                                       (short)0, acc01, false, false);
        acc10 = __builtin_amdgcn_wmma_f32_16x16x32_f16(false, af1, false, bf0,
                                                       (short)0, acc10, false, false);
        acc11 = __builtin_amdgcn_wmma_f32_16x16x32_f16(false, af1, false, bf1,
                                                       (short)0, acc11, false, false);
        __syncthreads();
    }

    // Epilogue: fuse bias/gain/(1-decay), transpose tile in LDS to [n][t]
    float* Ct = reinterpret_cast<float*>(smem); // [NT][CPITCH]
    {
        const int nl0 = wn * 32 + l16;
        const int nl1 = nl0 + 16;
        const float bz0 = bias[n0 + nl0];
        const float g0  = gain[n0 + nl0] * ONE_MINUS_DECAY_F;
        const float bz1 = bias[n0 + nl1];
        const float g1  = gain[n0 + nl1] * ONE_MINUS_DECAY_F;
        #pragma unroll
        for (int j = 0; j < 8; ++j) {
            const int tl0 = wm * 32 + j + 8 * h;      // frag rows 0..15
            const int tl1 = tl0 + 16;                 // frag rows 16..31
            Ct[nl0 * CPITCH + tl0] = (acc00[j] + bz0) * g0;
            Ct[nl1 * CPITCH + tl0] = (acc01[j] + bz1) * g1;
            Ct[nl0 * CPITCH + tl1] = (acc10[j] + bz0) * g0;
            Ct[nl1 * CPITCH + tl1] = (acc11[j] + bz1) * g1;
        }
    }
    __syncthreads();

    // Coalesced store: [b][n0+row][t0 + col], 128 rows x 16 float4
    float* outb = out + (size_t)b * N_DIM * T_DIM + (size_t)n0 * T_DIM + t0;
    #pragma unroll
    for (int it = 0; it < 8; ++it) {
        const int idx = it * 256 + tid;
        const int row = idx >> 4;
        const int col = (idx & 15) << 2;
        const float4 vv = *(const float4*)&Ct[row * CPITCH + col];
        *(float4*)(outb + (size_t)row * T_DIM + col) = vv;
    }
}

// Kernel 2: in-place leaky-integrate + threshold + reset-trace scan over T.
// Thread owns one (b,n) row of 128 contiguous floats in d_out: reads drive,
// overwrites with spikes, appends v and r tails.
__global__ __launch_bounds__(256) void snn_scan(float* __restrict__ out)
{
    const int bn = blockIdx.x * 256 + threadIdx.x; // 0 .. B*N-1
    float* row = out + (size_t)bn * T_DIM;
    float vmem = 0.0f, rst = 0.0f;
    for (int t4 = 0; t4 < T_DIM / 4; ++t4) {
        float4 d = *(float4*)(row + t4 * 4);
        vmem = vmem * DECAY_F + d.x; float s0 = (vmem > 1.0f) ? 1.0f : 0.0f; rst = rst * DECAY_F + s0; d.x = s0;
        vmem = vmem * DECAY_F + d.y; float s1 = (vmem > 1.0f) ? 1.0f : 0.0f; rst = rst * DECAY_F + s1; d.y = s1;
        vmem = vmem * DECAY_F + d.z; float s2 = (vmem > 1.0f) ? 1.0f : 0.0f; rst = rst * DECAY_F + s2; d.z = s2;
        vmem = vmem * DECAY_F + d.w; float s3 = (vmem > 1.0f) ? 1.0f : 0.0f; rst = rst * DECAY_F + s3; d.w = s3;
        *(float4*)(row + t4 * 4) = d;
    }
    const size_t tail = (size_t)B_DIM * N_DIM * T_DIM;
    out[tail + bn] = vmem;
    out[tail + (size_t)B_DIM * N_DIM + bn] = rst;
}

extern "C" void kernel_launch(void* const* d_in, const int* in_sizes, int n_in,
                              void* d_out, int out_size, void* d_ws, size_t ws_size,
                              hipStream_t stream) {
    (void)in_sizes; (void)n_in; (void)d_ws; (void)ws_size; (void)out_size;
    const float* x    = (const float*)d_in[0]; // [B, I, T]
    const float* W    = (const float*)d_in[1]; // [N, I]
    const float* bias = (const float*)d_in[2]; // [N]
    const float* gain = (const float*)d_in[3]; // [N]
    float* out = (float*)d_out;                // spikes [B,N,T] ++ v [B,N] ++ r [B,N]

    dim3 g1(N_DIM / NT, T_DIM / MT, B_DIM); // 16 x 2 x 128
    snn_gemm_drive<<<g1, 256, 0, stream>>>(x, W, bias, gain, out);

    snn_scan<<<(B_DIM * N_DIM) / 256, 256, 0, stream>>>(out);
}
